// behavior_model_69569880261223
// MI455X (gfx1250) — matrix operand order repaired and checked by reading, not compiled, not run on hardware
//
#include <hip/hip_runtime.h>

// ---------------- types ----------------
typedef __attribute__((ext_vector_type(16))) __bf16 v16bf;
typedef __attribute__((ext_vector_type(8)))  __bf16 v8bf;
typedef __attribute__((ext_vector_type(8)))  float  v8f;

union AFrag { v16bf v; v8bf h[2]; };

__device__ __forceinline__ __bf16 f2bf(float f) {
    unsigned u = __builtin_bit_cast(unsigned, f);
    unsigned r = (u + 0x7FFFu + ((u >> 16) & 1u)) >> 16;
    unsigned short s = (unsigned short)r;
    return __builtin_bit_cast(__bf16, s);
}

// Problem constants
#define F_ 32
#define B_ 6
#define NBOX 192      // F_*B_
#define C_ 1024
#define HF 14
#define NB 27
#define T_ 5
#define OC1 512
#define OC2 256
#define S1 45         // T_*3*3 output spatial positions of conv1a
#define MT32 6        // 192/32 M tiles (32 rows per wave)

// ---------------- ROI align: fmaps[32,1024,14,14], boxes[32,6,4] -> roi[fb][9][1024] bf16
__global__ __launch_bounds__(256) void roi_kernel(const float* __restrict__ fmaps,
                                                  const float* __restrict__ boxes,
                                                  __bf16* __restrict__ roi) {
    int fb = blockIdx.x;            // 0..191
    int f = fb / B_;
    int tid = threadIdx.x;

    __shared__ int   iy0[6], iy1[6], ix0[6], ix1[6];
    __shared__ float swy[6], swx[6];

    if (tid < 12) {
        int axis = tid / 6;         // 0 = y, 1 = x
        int gi = tid % 6;
        float g = (float)(gi >> 1) + ((float)(gi & 1) + 0.5f) * 0.5f;
        float c1 = boxes[(size_t)fb * 4 + (axis == 0 ? 1 : 0)];
        float c2 = boxes[(size_t)fb * 4 + (axis == 0 ? 3 : 2)];
        float bsc = fmaxf(c2 - c1, 1.0f) * (1.0f / 3.0f);
        float pos = c1 + bsc * g;
        float p0f = floorf(pos);
        p0f = fminf(fmaxf(p0f, 0.0f), (float)(HF - 1));
        float l = fminf(fmaxf(pos - p0f, 0.0f), 1.0f);
        int p0 = (int)p0f;
        int p1 = min(p0 + 1, HF - 1);
        if (axis == 0) { iy0[gi] = p0; iy1[gi] = p1; swy[gi] = l; }
        else           { ix0[gi] = p0; ix1[gi] = p1; swx[gi] = l; }
    }
    __syncthreads();

    for (int c = tid; c < C_; c += 256) {
        const float* Fm = fmaps + ((size_t)f * C_ + c) * (HF * HF);
        #pragma unroll
        for (int p = 0; p < 9; ++p) {
            int oyy = p / 3, oxx = p % 3;
            float sum = 0.0f;
            #pragma unroll
            for (int sy = 0; sy < 2; ++sy) {
                int gy = oyy * 2 + sy;
                float ly = swy[gy];
                int y0 = iy0[gy], y1 = iy1[gy];
                #pragma unroll
                for (int sx = 0; sx < 2; ++sx) {
                    int gx = oxx * 2 + sx;
                    float lx = swx[gx];
                    int x0 = ix0[gx], x1 = ix1[gx];
                    float v = Fm[y0 * HF + x0] * (1.0f - ly) * (1.0f - lx)
                            + Fm[y0 * HF + x1] * (1.0f - ly) * lx
                            + Fm[y1 * HF + x0] * ly * (1.0f - lx)
                            + Fm[y1 * HF + x1] * ly * lx;
                    sum += v;
                }
            }
            roi[((size_t)fb * 9 + p) * C_ + c] = f2bf(sum * 0.25f);
        }
    }
}

// ---------------- temporal gather: roi -> Xg[pos(45)][n(192)][c(1024)] bf16 (zeros where absent)
__global__ __launch_bounds__(128) void gather_kernel(const __bf16* __restrict__ roi,
                                                     const int* __restrict__ pid,
                                                     __bf16* __restrict__ Xg) {
    int posn = blockIdx.x;          // 0 .. 45*192-1
    int pos = posn / NBOX;
    int n = posn % NBOX;
    int t = pos / 9, sp = pos % 9;
    int f = n / B_, b = n % B_;
    int p = pid[f * B_ + b];
    int start = max(f - 2, 0);
    int end = min(start + 4, F_ - 1);
    int g = (int)((float)start + (float)t * (float)(end - start) * 0.25f);
    int j = -1;
    #pragma unroll
    for (int j2 = 0; j2 < B_; ++j2)
        if (pid[g * B_ + j2] == p) { j = j2; break; }

    uint4* dst = (uint4*)(Xg + ((size_t)pos * NBOX + n) * C_);
    if (j >= 0) {
        const uint4* src = (const uint4*)(roi + (((size_t)g * B_ + j) * 9 + sp) * C_);
        dst[threadIdx.x] = src[threadIdx.x];
    } else {
        dst[threadIdx.x] = make_uint4(0u, 0u, 0u, 0u);
    }
}

// ---------------- weight transforms: [oc][ic][27] f32 -> [tap][oc][ic] bf16
__global__ __launch_bounds__(256) void wt1_kernel(const float* __restrict__ w, __bf16* __restrict__ Wb) {
    size_t i = (size_t)blockIdx.x * 256 + threadIdx.x;   // 27 << 19 total
    int tap = (int)(i >> 19);
    int oc = (int)((i >> 10) & 511);
    int ic = (int)(i & 1023);
    Wb[i] = f2bf(w[((size_t)oc * C_ + ic) * 27 + tap]);
}

__global__ __launch_bounds__(256) void wt2_kernel(const float* __restrict__ w, __bf16* __restrict__ Wb) {
    size_t i = (size_t)blockIdx.x * 256 + threadIdx.x;   // 27 << 17 total
    int tap = (int)(i >> 17);
    int oc = (int)((i >> 9) & 255);
    int ic = (int)(i & 511);
    Wb[i] = f2bf(w[((size_t)oc * OC1 + ic) * 27 + tap]);
}

// ---------------- conv1a as WMMA GEMM: Xg x Wb1 -> Y1p[pos(45)][n(192)][oc(512)] bf16 (bias+ReLU)
// per-wave tile: 32 rows (n) x 64 cols (oc); taps masked uniformly per output position
__global__ __launch_bounds__(256) void conv1a_wmma(const __bf16* __restrict__ Xg,
                                                   const __bf16* __restrict__ Wb,
                                                   const float* __restrict__ bias,
                                                   __bf16* __restrict__ Y) {
    int lane = threadIdx.x & 31;
    int wave = blockIdx.x * 8 + (threadIdx.x >> 5);   // 45*6*8 = 2160 waves total
    int ng = wave & 7;                                // 8 groups of 64 output cols
    int mt = (wave >> 3) % MT32;                      // 6 tiles of 32 rows
    int s  = wave / 48;                               // 0..44
    int ot = s / 9, oy = (s % 9) / 3, ox = s % 3;
    int l15 = lane & 15;
    bool low = lane < 16;
    int aoff = low ? 0 : 16;
    int boff = low ? 0 : 16;

    v8f acc[2][4] = {};

    for (int kt = 0; kt < 3; ++kt) {
        int it = ot + kt - 1;
        if ((unsigned)it >= 5u) continue;
        for (int ky = 0; ky < 3; ++ky) {
            int iy = oy + ky - 1;
            if ((unsigned)iy >= 3u) continue;
            for (int kx = 0; kx < 3; ++kx) {
                int ix = ox + kx - 1;
                if ((unsigned)ix >= 3u) continue;
                int pos = (it * 3 + iy) * 3 + ix;
                int tap = (kt * 3 + ky) * 3 + kx;
                const __bf16* ap = Xg + (((size_t)pos * NBOX) + (size_t)mt * 32 + l15) * C_ + aoff;
                const __bf16* bp = Wb + (((size_t)tap * OC1) + (size_t)ng * 64 + l15) * C_ + boff;
                #pragma unroll 2
                for (int cc = 0; cc < C_; cc += 32) {
                    AFrag a0, a1;
                    a0.h[0] = *(const v8bf*)(ap + cc);
                    a0.h[1] = *(const v8bf*)(ap + cc + 8);
                    a1.h[0] = *(const v8bf*)(ap + 16 * C_ + cc);
                    a1.h[1] = *(const v8bf*)(ap + 16 * C_ + cc + 8);
                    #pragma unroll
                    for (int j = 0; j < 4; ++j) {
                        v16bf b = *(const v16bf*)(bp + j * 16 * C_ + cc);
                        acc[0][j] = __builtin_amdgcn_wmma_f32_16x16x32_bf16(false, a0.v, false, b, (short)0, acc[0][j], false, false);
                        acc[1][j] = __builtin_amdgcn_wmma_f32_16x16x32_bf16(false, a1.v, false, b, (short)0, acc[1][j], false, false);
                    }
                }
            }
        }
    }

    int col = l15;
    #pragma unroll
    for (int j = 0; j < 4; ++j) {
        int oc = ng * 64 + j * 16 + col;
        float bv = bias[oc];
        #pragma unroll
        for (int i = 0; i < 2; ++i) {
            #pragma unroll
            for (int r = 0; r < 8; ++r) {
                int m = r + (low ? 0 : 8);
                int row = mt * 32 + i * 16 + m;
                float v = acc[i][j][r] + bv;
                v = v > 0.0f ? v : 0.0f;
                Y[((size_t)s * NBOX + row) * OC1 + oc] = f2bf(v);
            }
        }
    }
}

// ---------------- conv1b as WMMA GEMM: Y1p x Wb2 -> Y2[ot'(3)][n(192)][oc(256)] f32 (bias+ReLU)
__global__ __launch_bounds__(256) void conv1b_wmma(const __bf16* __restrict__ Y1p,
                                                   const __bf16* __restrict__ Wb,
                                                   const float* __restrict__ bias,
                                                   float* __restrict__ Y2) {
    int lane = threadIdx.x & 31;
    int wave = blockIdx.x * 8 + (threadIdx.x >> 5);   // 3*6*4 = 72 waves total
    int ng = wave & 3;                                // 4 groups of 64 output cols
    int mt = (wave >> 2) % MT32;                      // 6 tiles of 32 rows
    int otp = wave / 24;                              // 0..2
    int l15 = lane & 15;
    bool low = lane < 16;
    int aoff = low ? 0 : 16;
    int boff = low ? 0 : 16;

    v8f acc[2][4] = {};

    for (int kt = 0; kt < 3; ++kt) {
        for (int ky = 0; ky < 3; ++ky) {
            for (int kx = 0; kx < 3; ++kx) {
                int pos = ((otp + kt) * 3 + ky) * 3 + kx;
                int tap = (kt * 3 + ky) * 3 + kx;
                const __bf16* ap = Y1p + (((size_t)pos * NBOX) + (size_t)mt * 32 + l15) * OC1 + aoff;
                const __bf16* bp = Wb + (((size_t)tap * OC2) + (size_t)ng * 64 + l15) * OC1 + boff;
                #pragma unroll 2
                for (int cc = 0; cc < OC1; cc += 32) {
                    AFrag a0, a1;
                    a0.h[0] = *(const v8bf*)(ap + cc);
                    a0.h[1] = *(const v8bf*)(ap + cc + 8);
                    a1.h[0] = *(const v8bf*)(ap + 16 * OC1 + cc);
                    a1.h[1] = *(const v8bf*)(ap + 16 * OC1 + cc + 8);
                    #pragma unroll
                    for (int j = 0; j < 4; ++j) {
                        v16bf b = *(const v16bf*)(bp + j * 16 * OC1 + cc);
                        acc[0][j] = __builtin_amdgcn_wmma_f32_16x16x32_bf16(false, a0.v, false, b, (short)0, acc[0][j], false, false);
                        acc[1][j] = __builtin_amdgcn_wmma_f32_16x16x32_bf16(false, a1.v, false, b, (short)0, acc[1][j], false, false);
                    }
                }
            }
        }
    }

    int col = l15;
    #pragma unroll
    for (int j = 0; j < 4; ++j) {
        int oc = ng * 64 + j * 16 + col;
        float bv = bias[oc];
        #pragma unroll
        for (int i = 0; i < 2; ++i) {
            #pragma unroll
            for (int r = 0; r < 8; ++r) {
                int m = r + (low ? 0 : 8);
                int row = mt * 32 + i * 16 + m;
                float v = acc[i][j][r] + bv;
                v = v > 0.0f ? v : 0.0f;
                Y2[((size_t)otp * NBOX + row) * OC2 + oc] = v;
            }
        }
    }
}

// ---------------- max over T + FC head
__global__ __launch_bounds__(256) void head_kernel(const float* __restrict__ Y2,
                                                   const float* __restrict__ fcw,
                                                   const float* __restrict__ fcb,
                                                   float* __restrict__ out) {
    int n = blockIdx.x;             // 0..191
    int tid = threadIdx.x;          // 256
    __shared__ float xm[OC2];
    float a0 = Y2[((size_t)0 * NBOX + n) * OC2 + tid];
    float a1 = Y2[((size_t)1 * NBOX + n) * OC2 + tid];
    float a2 = Y2[((size_t)2 * NBOX + n) * OC2 + tid];
    xm[tid] = fmaxf(fmaxf(a0, a1), a2);
    __syncthreads();
    if (tid < NB) {
        float s = fcb[tid];
        const float* wr = fcw + (size_t)tid * OC2;
        #pragma unroll 8
        for (int i = 0; i < OC2; ++i) s += wr[i] * xm[i];
        out[(size_t)n * NB + tid] = s;
    }
}

// ---------------- launcher ----------------
extern "C" void kernel_launch(void* const* d_in, const int* in_sizes, int n_in,
                              void* d_out, int out_size, void* d_ws, size_t ws_size,
                              hipStream_t stream) {
    const float* fmaps = (const float*)d_in[0];
    const float* boxes = (const float*)d_in[1];
    const int*   pid   = (const int*)d_in[2];
    const float* w1    = (const float*)d_in[3];
    const float* b1    = (const float*)d_in[4];
    const float* w2    = (const float*)d_in[5];
    const float* b2    = (const float*)d_in[6];
    const float* fcw   = (const float*)d_in[7];
    const float* fcb   = (const float*)d_in[8];
    float* out = (float*)d_out;

    char* ws = (char*)d_ws;
    size_t off = 0;
    auto carve = [&](size_t bytes) -> void* {
        void* p = ws + off;
        off += (bytes + 255) & ~(size_t)255;
        return p;
    };

    __bf16* Xg  = (__bf16*)carve((size_t)S1 * NBOX * C_ * 2);          // 17.7 MB
    __bf16* roi = (__bf16*)carve((size_t)NBOX * 9 * C_ * 2);           //  3.5 MB
    __bf16* Wb1 = (__bf16*)carve((size_t)27 * OC1 * C_ * 2);           // 28.3 MB
    __bf16* Wb2 = (__bf16*)carve((size_t)27 * OC2 * OC1 * 2);          //  7.1 MB
    __bf16* Y1p = (__bf16*)carve((size_t)S1 * NBOX * OC1 * 2);         //  8.8 MB
    float*  Y2  = (float*)carve((size_t)3 * NBOX * OC2 * 4);           //  0.6 MB
    (void)ws_size; (void)in_sizes; (void)n_in; (void)out_size;

    roi_kernel<<<NBOX, 256, 0, stream>>>(fmaps, boxes, roi);
    gather_kernel<<<S1 * NBOX, 128, 0, stream>>>(roi, pid, Xg);
    wt1_kernel<<<(27u << 19) / 256, 256, 0, stream>>>(w1, Wb1);
    wt2_kernel<<<(27u << 17) / 256, 256, 0, stream>>>(w2, Wb2);
    conv1a_wmma<<<270, 256, 0, stream>>>(Xg, Wb1, b1, Y1p);            // 45*6*8 waves
    conv1b_wmma<<<9, 256, 0, stream>>>(Y1p, Wb2, b2, Y2);              // 3*6*4 waves
    head_kernel<<<NBOX, 256, 0, stream>>>(Y2, fcw, fcb, out);
}
